// MyNetwork_50568944943353
// MI455X (gfx1250) — matrix operand order repaired and checked by reading, not compiled, not run on hardware
//
#include <hip/hip_runtime.h>

#define DEV __device__ __forceinline__

typedef __attribute__((ext_vector_type(16))) __bf16 v16bf;
typedef __attribute__((ext_vector_type(8)))  __bf16 v8bf;
typedef __attribute__((ext_vector_type(8)))  float  v8f;
typedef __attribute__((ext_vector_type(4)))  unsigned int u32x4;
typedef __attribute__((ext_vector_type(8)))  int i32x8;
typedef __attribute__((ext_vector_type(4)))  int i32x4;

static constexpr int Bq  = 16;
static constexpr int Sq  = 1024;
static constexpr int Dq  = 512;
static constexpr int Hq  = 4;
static constexpr int FFq = 2048;
static constexpr int DHq = Dq / Hq;      // 128
static constexpr int Mrows = Bq * Sq;    // 16384

#if defined(__AMDGCN__) && __has_builtin(__builtin_amdgcn_tensor_load_to_lds) && \
    __has_builtin(__builtin_amdgcn_s_wait_tensorcnt)
#define USE_TDM 1
#else
#define USE_TDM 0
#endif

struct Pair8 { v8bf lo; v8bf hi; };
union  FragU { v16bf v; Pair8 p; };

// ---- fragment loaders (CDNA5 WMMA VGPR layouts, ISA 7.12.2) ----
// A-matrix 16x32 bf16: lane r=l&15 holds row M=r; K = c0..c0+7 and c0+16..c0+23
// with c0 = (l>=16)*8  -> two contiguous 16B loads per lane.
DEV v16bf load_fragA(const __bf16* base, int ld, int lane) {
  int r = lane & 15, g = lane >> 4;
  const __bf16* p = base + (size_t)r * ld + g * 8;
  FragU u;
  u.p.lo = *(const v8bf*)(p);
  u.p.hi = *(const v8bf*)(p + 16);
  return u.v;
}
// B-matrix 32x16 bf16: lane r=l&15 holds column N=r; K = c0..c0+7 and
// c0+16..c0+23 with c0 = (l>=16)*8 -> two contiguous 16B loads per lane from W^T rows.
DEV v16bf load_fragB(const __bf16* base, int ld, int lane) {
  int r = lane & 15, g = lane >> 4;
  const __bf16* p = base + (size_t)r * ld + g * 8;
  FragU u;
  u.p.lo = *(const v8bf*)(p);
  u.p.hi = *(const v8bf*)(p + 16);
  return u.v;
}

DEV v8f wmma_bf16(v16bf a, v16bf b, v8f c) {
  return __builtin_amdgcn_wmma_f32_16x16x32_bf16(false, a, false, b, (short)0, c,
                                                 false, false);
}

#if USE_TDM
// ---- Tensor Data Mover: 2D bf16 tile (tileW x tileH) -> LDS ----
// Descriptor packing per CDNA5 ISA 8.3/8.4: group0 {count=1, lds_addr,
// global_addr, type=2}; group1 {data_size=2B, tensor dims, tile dims, stride}.
// This toolchain's builtin takes 6 args (4 D# SGPR groups + spare + cpol).
DEV void tdm_load_tile(const void* g, unsigned ldsoff, int tileW, int tileH,
                       int strideElts) {
  unsigned long long ga = (unsigned long long)(uintptr_t)g;
  unsigned galo = (unsigned)__builtin_amdgcn_readfirstlane((int)(unsigned)(ga & 0xffffffffu));
  unsigned gahi = (unsigned)__builtin_amdgcn_readfirstlane((int)(unsigned)(ga >> 32));
  unsigned lo   = (unsigned)__builtin_amdgcn_readfirstlane((int)ldsoff);
  unsigned td0 = (unsigned)tileW, td1 = (unsigned)tileH;
  unsigned long long st = (unsigned long long)(unsigned)strideElts;

  u32x4 g0;
  g0[0] = 1u;                                  // count=1 (valid user descriptor)
  g0[1] = lo;                                  // lds_addr (bytes)
  g0[2] = galo;                                // global_addr[31:0]
  g0[3] = (gahi & 0x1ffffffu) | (2u << 30);    // global_addr[56:32] | type=2

  i32x8 g1;
  g1[0] = (int)(1u << 16);                                       // data_size=2B
  g1[1] = (int)((td0 & 0xffffu) << 16);                          // tensor_dim0 lo
  g1[2] = (int)(((td0 >> 16) & 0xffffu) | ((td1 & 0xffffu) << 16)); // td0 hi | td1 lo
  g1[3] = (int)(((td1 >> 16) & 0xffffu) | ((td0 & 0xffffu) << 16)); // td1 hi | tile_dim0
  g1[4] = (int)(td1 & 0xffffu);                                  // tile_dim1 (tile_dim2=0)
  g1[5] = (int)(unsigned)(st & 0xffffffffu);                     // dim0 stride lo
  g1[6] = (int)(unsigned)((st >> 32) & 0xffffu);                 // dim0 stride hi
  g1[7] = 0;
  i32x4 gz  = {0, 0, 0, 0};
  i32x8 gz8 = {0, 0, 0, 0, 0, 0, 0, 0};
  __builtin_amdgcn_tensor_load_to_lds(g0, g1, gz, gz, gz8, 0);
}
#endif

// ---------------- weight transpose + f32->bf16 convert ----------------
__global__ void tconv_kernel(const float* __restrict__ w, __bf16* __restrict__ wt,
                             int K, int N) {
  int i = blockIdx.x * 256 + threadIdx.x;
  if (i >= K * N) return;
  int k = i / N, n = i % N;
  wt[(size_t)n * K + k] = (__bf16)w[i];
}

// ---------------- embedding gather + positional encoding ----------------
__global__ void embed_kernel(const int* __restrict__ inp, const float* __restrict__ emb,
                             const float* __restrict__ pe, float* __restrict__ xf,
                             __bf16* __restrict__ xh) {
  int i = blockIdx.x * 256 + threadIdx.x;          // grid sized exactly
  int row = i >> 9;                                // / Dq (=512)
  int col = i & (Dq - 1);
  int s   = row & (Sq - 1);
  int tok = inp[row];
  float v = emb[(size_t)tok * Dq + col] + pe[(size_t)s * Dq + col];
  xf[i] = v;
  xh[i] = (__bf16)v;
}

// ---------------- WMMA GEMM: C = relu(A @ Wt^T + bias) ----------------
// A [M,K] bf16 row-major, Wt [N,K] bf16 row-major (pre-transposed weights).
// 64x64 per wave: 4 A-frags x 4 B-frags -> 16 WMMAs per 32-wide K step.
// OUT_MODE 0: bf16 [M,N]; 1: bf16 V-transposed to [B,H,DH,S]; 2: f32 [M,N].
template <int OUT_MODE>
__global__ __launch_bounds__(128) void gemm_wmma_kernel(
    const __bf16* __restrict__ A, const __bf16* __restrict__ Wt,
    const float* __restrict__ bias, __bf16* __restrict__ outh,
    float* __restrict__ outf, int M, int N, int K) {
  int lane = threadIdx.x & 31;
  int wave = threadIdx.x >> 5;
  int row0 = blockIdx.y * 128 + (wave >> 1) * 64;
  int col0 = blockIdx.x * 128 + (wave & 1) * 64;
  int r = lane & 15, g = lane >> 4;

  v8f acc[4][4];
  v8f zero = {};
#pragma unroll
  for (int i = 0; i < 4; ++i)
#pragma unroll
    for (int j = 0; j < 4; ++j) acc[i][j] = zero;

  for (int kk = 0; kk < K; kk += 32) {
    v16bf af[4], bf[4];
#pragma unroll
    for (int i = 0; i < 4; ++i)
      af[i] = load_fragA(A + (size_t)(row0 + 16 * i) * K + kk, K, lane);
#pragma unroll
    for (int j = 0; j < 4; ++j)
      bf[j] = load_fragB(Wt + (size_t)(col0 + 16 * j) * K + kk, K, lane);
    if (kk + 32 < K) {   // hint next K-tile into cache (global_prefetch_b8)
      __builtin_prefetch(A + (size_t)(row0 + r) * K + kk + 32, 0, 1);
      __builtin_prefetch(Wt + (size_t)(col0 + r) * K + kk + 32, 0, 1);
    }
#pragma unroll
    for (int i = 0; i < 4; ++i)
#pragma unroll
      for (int j = 0; j < 4; ++j)
        acc[i][j] = wmma_bf16(af[i], bf[j], acc[i][j]);
  }

#pragma unroll
  for (int i = 0; i < 4; ++i) {
#pragma unroll
    for (int j = 0; j < 4; ++j) {
      int n = col0 + j * 16 + r;
      float bv = bias[n];
#pragma unroll
      for (int jj = 0; jj < 8; ++jj) {
        int m = row0 + i * 16 + g * 8 + jj;  // C layout: VGPR jj -> row jj (+8 hi)
        float v = acc[i][j][jj] + bv;
        v = fmaxf(v, 0.0f);
        if (OUT_MODE == 2) {
          outf[(size_t)m * N + n] = v;
        } else if (OUT_MODE == 0) {
          outh[(size_t)m * N + n] = (__bf16)v;
        } else {
          // scatter V head-transposed: [B,H,DH,S]
          int b_ = m >> 10, s_ = m & (Sq - 1);
          int h_ = n >> 7,  d_ = n & (DHq - 1);
          outh[(size_t)((b_ * Hq + h_) * DHq + d_) * Sq + s_] = (__bf16)v;
        }
      }
    }
  }
}

// ---------------- flash attention: one wave per 16-query tile ----------------
// 4 waves/block share (b,h); K chunks staged to LDS by the Tensor Data Mover
// (double buffered, s_wait_tensorcnt), V read via contiguous B-frags from V^T.
__global__ __launch_bounds__(128) void attention_kernel(
    const __bf16* __restrict__ qb, const __bf16* __restrict__ kb,
    const __bf16* __restrict__ vtb, __bf16* __restrict__ yh) {
  __shared__ alignas(32) __bf16 Plds[4][16 * 32];
#if USE_TDM
  __shared__ alignas(64) __bf16 Kst[2][32 * DHq];   // shared K chunk, 2 x 8KB
#endif
  int lane = threadIdx.x & 31;
  int wave = threadIdx.x >> 5;
  int t  = blockIdx.x * 4 + wave;
  int qt = t & 63;            // S/16 = 64 query tiles per (b,h)
  int bh = t >> 6;
  int h  = bh & (Hq - 1);
  int b  = bh >> 2;
  int r = lane & 15, g = lane >> 4;

  size_t qrow = (size_t)b * Sq + qt * 16;
  const __bf16* qp = qb  + qrow * Dq + h * DHq;
  const __bf16* kp = kb  + (size_t)b * Sq * Dq + h * DHq;
  const __bf16* vp = vtb + (size_t)(b * Hq + h) * DHq * Sq;   // [DH, S]
  __bf16* Pw = &Plds[wave][0];

  v16bf qa[4];
#pragma unroll
  for (int kk = 0; kk < 4; ++kk) qa[kk] = load_fragA(qp + kk * 32, Dq, lane);

  float Mr[8], Lr[8];
  v8f acc[8];
  v8f zero = {};
#pragma unroll
  for (int j = 0; j < 8; ++j) { Mr[j] = -3.0e38f; Lr[j] = 0.f; }
#pragma unroll
  for (int tn = 0; tn < 8; ++tn) acc[tn] = zero;

  const float scale = 0.08838834764831845f;  // 1/sqrt(DH=128)

#if USE_TDM
  if (wave == 0)
    tdm_load_tile(kp, (unsigned)(uintptr_t)&Kst[0][0], DHq, 32, Dq);
#endif

  for (int c = 0; c < Sq / 32; ++c) {
    int kv0 = c * 32;
#if USE_TDM
    __syncthreads();                 // all readers of buf[(c+1)&1] finished
    if (wave == 0) {
      if (c + 1 < Sq / 32) {
        tdm_load_tile(kp + (size_t)(kv0 + 32) * Dq,
                      (unsigned)(uintptr_t)&Kst[(c + 1) & 1][0], DHq, 32, Dq);
        __builtin_amdgcn_s_wait_tensorcnt(1);   // chunk c landed in LDS
      } else {
        __builtin_amdgcn_s_wait_tensorcnt(0);
      }
    }
    __syncthreads();                 // chunk c visible to all waves
    const __bf16* kl = &Kst[c & 1][0];
#endif
    // scores S[16q x 32kv] = Q (16x128) * K^T : two 16x16 D-tiles
    v8f s0 = zero, s1 = zero;
#pragma unroll
    for (int kk = 0; kk < 4; ++kk) {
#if USE_TDM
      v16bf kb0 = load_fragB(kl + kk * 32, DHq, lane);
      v16bf kb1 = load_fragB(kl + (size_t)16 * DHq + kk * 32, DHq, lane);
#else
      v16bf kb0 = load_fragB(kp + (size_t)kv0 * Dq + kk * 32, Dq, lane);
      v16bf kb1 = load_fragB(kp + (size_t)(kv0 + 16) * Dq + kk * 32, Dq, lane);
#endif
      s0 = wmma_bf16(qa[kk], kb0, s0);
      s1 = wmma_bf16(qa[kk], kb1, s1);
    }
    // online softmax; C-layout: VGPR j -> row j + 8*g, lanes&15 -> columns
#pragma unroll
    for (int j = 0; j < 8; ++j) {
      float a0 = s0[j] * scale;
      float a1 = s1[j] * scale;
      float mx = fmaxf(a0, a1);
      mx = fmaxf(mx, __shfl_xor(mx, 1, 32));
      mx = fmaxf(mx, __shfl_xor(mx, 2, 32));
      mx = fmaxf(mx, __shfl_xor(mx, 4, 32));
      mx = fmaxf(mx, __shfl_xor(mx, 8, 32));
      float Mn = fmaxf(Mr[j], mx);
      float corr = __expf(Mr[j] - Mn);
      Mr[j] = Mn;
      float p0 = __expf(a0 - Mn);
      float p1 = __expf(a1 - Mn);
      float rs = p0 + p1;
      rs += __shfl_xor(rs, 1, 32);
      rs += __shfl_xor(rs, 2, 32);
      rs += __shfl_xor(rs, 4, 32);
      rs += __shfl_xor(rs, 8, 32);
      Lr[j] = Lr[j] * corr + rs;
#pragma unroll
      for (int tn = 0; tn < 8; ++tn) acc[tn][j] *= corr;
      int m = g * 8 + j;
      Pw[m * 32 + r]      = (__bf16)p0;   // P tile (16x32) -> LDS, row-major
      Pw[m * 32 + r + 16] = (__bf16)p1;
    }
    __builtin_amdgcn_wave_barrier();
#if __has_builtin(__builtin_amdgcn_s_wait_dscnt)
    __builtin_amdgcn_s_wait_dscnt(0);     // P stores visible before A-frag reload
#endif
    // re-layout P as A-fragment from LDS (two ds_load_b128 per lane)
    v16bf pa = load_fragA(Pw, 32, lane);
    // O += P (16x32) * V (32x128): 8 N-tiles, V^T gives contiguous B-frags
#pragma unroll
    for (int tn = 0; tn < 8; ++tn) {
      v16bf vb = load_fragB(vp + (size_t)(tn * 16) * Sq + kv0, Sq, lane);
      acc[tn] = wmma_bf16(pa, vb, acc[tn]);
    }
  }

  __bf16* yp = yh + qrow * Dq + h * DHq;
#pragma unroll
  for (int tn = 0; tn < 8; ++tn) {
#pragma unroll
    for (int j = 0; j < 8; ++j) {
      int m = g * 8 + j;
      float o = acc[tn][j] / Lr[j];
      yp[(size_t)m * Dq + tn * 16 + r] = (__bf16)o;
    }
  }
}

// ---------------- residual + LayerNorm (one wave per row) ----------------
__global__ __launch_bounds__(256) void ln_kernel(
    const float* __restrict__ xf, const float* __restrict__ ho,
    const float* __restrict__ gam, const float* __restrict__ bet,
    __bf16* __restrict__ hn) {
  int gid  = blockIdx.x * blockDim.x + threadIdx.x;
  int row  = gid >> 5;
  int lane = gid & 31;
  if (row >= Mrows) return;
  const float* px = xf + (size_t)row * Dq;
  const float* py = ho + (size_t)row * Dq;
  float hv[16];
  float s = 0.f;
#pragma unroll
  for (int i = 0; i < 16; ++i) {
    int col = i * 32 + lane;
    hv[i] = px[col] + py[col];
    s += hv[i];
  }
  s += __shfl_xor(s, 16, 32); s += __shfl_xor(s, 8, 32);
  s += __shfl_xor(s, 4, 32);  s += __shfl_xor(s, 2, 32);
  s += __shfl_xor(s, 1, 32);
  float mu = s * (1.0f / Dq);
  float v = 0.f;
#pragma unroll
  for (int i = 0; i < 16; ++i) { float d = hv[i] - mu; v += d * d; }
  v += __shfl_xor(v, 16, 32); v += __shfl_xor(v, 8, 32);
  v += __shfl_xor(v, 4, 32);  v += __shfl_xor(v, 2, 32);
  v += __shfl_xor(v, 1, 32);
  float rstd = rsqrtf(v * (1.0f / Dq) + 1e-5f);
  __bf16* po = hn + (size_t)row * Dq;
#pragma unroll
  for (int i = 0; i < 16; ++i) {
    int col = i * 32 + lane;
    po[col] = (__bf16)((hv[i] - mu) * rstd * gam[col] + bet[col]);
  }
}

// ---------------- FFN2: [M,2048] @ [2048,2] + b2 (one wave per row) ----------------
__global__ __launch_bounds__(256) void ffn2_kernel(
    const __bf16* __restrict__ fh, const float* __restrict__ w2,
    const float* __restrict__ b2, float* __restrict__ out) {
  int gid  = blockIdx.x * blockDim.x + threadIdx.x;
  int row  = gid >> 5;
  int lane = gid & 31;
  if (row >= Mrows) return;
  const __bf16* pf = fh + (size_t)row * FFq;
  float a0 = 0.f, a1 = 0.f;
  for (int k = lane; k < FFq; k += 32) {
    float f = (float)pf[k];
    a0 += f * w2[2 * k];
    a1 += f * w2[2 * k + 1];
  }
  a0 += __shfl_xor(a0, 16, 32); a0 += __shfl_xor(a0, 8, 32);
  a0 += __shfl_xor(a0, 4, 32);  a0 += __shfl_xor(a0, 2, 32);
  a0 += __shfl_xor(a0, 1, 32);
  a1 += __shfl_xor(a1, 16, 32); a1 += __shfl_xor(a1, 8, 32);
  a1 += __shfl_xor(a1, 4, 32);  a1 += __shfl_xor(a1, 2, 32);
  a1 += __shfl_xor(a1, 1, 32);
  if (lane == 0) {
    out[2 * row]     = a0 + b2[0];
    out[2 * row + 1] = a1 + b2[1];
  }
}

// ---------------- launcher ----------------
extern "C" void kernel_launch(void* const* d_in, const int* in_sizes, int n_in,
                              void* d_out, int out_size, void* d_ws, size_t ws_size,
                              hipStream_t stream) {
  (void)in_sizes; (void)n_in; (void)out_size; (void)ws_size;
  const int*   inp = (const int*)  d_in[0];
  const float* emb = (const float*)d_in[1];
  const float* wq  = (const float*)d_in[2];
  const float* bq  = (const float*)d_in[3];
  const float* wk  = (const float*)d_in[4];
  const float* bk  = (const float*)d_in[5];
  const float* wv  = (const float*)d_in[6];
  const float* bv  = (const float*)d_in[7];
  const float* wo  = (const float*)d_in[8];
  const float* bo  = (const float*)d_in[9];
  const float* lng = (const float*)d_in[10];
  const float* lnb = (const float*)d_in[11];
  const float* w1  = (const float*)d_in[12];
  const float* b1  = (const float*)d_in[13];
  const float* w2  = (const float*)d_in[14];
  const float* b2  = (const float*)d_in[15];
  const float* pe  = (const float*)d_in[16];
  float* out = (float*)d_out;

  char* p = (char*)d_ws;
  auto alloc = [&](size_t bytes) {
    char* q = p;
    p += (bytes + 255) & ~(size_t)255;
    return q;
  };
  float*  xf  = (float*) alloc((size_t)Mrows * Dq * 4);   // x (f32, residual)
  __bf16* xh  = (__bf16*)alloc((size_t)Mrows * Dq * 2);   // x (bf16, GEMM A)
  __bf16* qhb = (__bf16*)alloc((size_t)Mrows * Dq * 2);
  __bf16* khb = (__bf16*)alloc((size_t)Mrows * Dq * 2);
  __bf16* vtb = (__bf16*)alloc((size_t)Mrows * Dq * 2);   // V as [B,H,DH,S]
  __bf16* yhb = (__bf16*)alloc((size_t)Mrows * Dq * 2);   // attention out
  float*  ho  = (float*) alloc((size_t)Mrows * Dq * 4);   // relu(y@wo+bo)
  __bf16* hn  = (__bf16*)alloc((size_t)Mrows * Dq * 2);   // LN out
  __bf16* fh  = (__bf16*)alloc((size_t)Mrows * FFq * 2);  // relu(hn@w1+b1)
  __bf16* wqt = (__bf16*)alloc((size_t)Dq * Dq * 2);
  __bf16* wkt = (__bf16*)alloc((size_t)Dq * Dq * 2);
  __bf16* wvt = (__bf16*)alloc((size_t)Dq * Dq * 2);
  __bf16* wot = (__bf16*)alloc((size_t)Dq * Dq * 2);
  __bf16* w1t = (__bf16*)alloc((size_t)Dq * FFq * 2);

  // weight convert + transpose (bf16, [N,K])
  int wdd = (Dq * Dq + 255) / 256;
  tconv_kernel<<<wdd, 256, 0, stream>>>(wq, wqt, Dq, Dq);
  tconv_kernel<<<wdd, 256, 0, stream>>>(wk, wkt, Dq, Dq);
  tconv_kernel<<<wdd, 256, 0, stream>>>(wv, wvt, Dq, Dq);
  tconv_kernel<<<wdd, 256, 0, stream>>>(wo, wot, Dq, Dq);
  tconv_kernel<<<(Dq * FFq + 255) / 256, 256, 0, stream>>>(w1, w1t, Dq, FFq);

  // x = emb[inputs] + pe
  embed_kernel<<<(Mrows * Dq) / 256, 256, 0, stream>>>(inp, emb, pe, xf, xh);

  // Q,K,V projections (relu), V written head-transposed
  dim3 gQ(Dq / 128, Mrows / 128);
  gemm_wmma_kernel<0><<<gQ, 128, 0, stream>>>(xh, wqt, bq, qhb, nullptr, Mrows, Dq, Dq);
  gemm_wmma_kernel<0><<<gQ, 128, 0, stream>>>(xh, wkt, bk, khb, nullptr, Mrows, Dq, Dq);
  gemm_wmma_kernel<1><<<gQ, 128, 0, stream>>>(xh, wvt, bv, vtb, nullptr, Mrows, Dq, Dq);

  // flash attention -> y
  attention_kernel<<<(Bq * Hq * (Sq / 16)) / 4, 128, 0, stream>>>(qhb, khb, vtb, yhb);

  // relu(y @ wo + bo) in f32 for residual
  gemm_wmma_kernel<2><<<gQ, 128, 0, stream>>>(yhb, wot, bo, nullptr, ho, Mrows, Dq, Dq);

  // residual + LayerNorm -> hn (bf16)
  ln_kernel<<<Mrows / 8, 256, 0, stream>>>(xf, ho, lng, lnb, hn);

  // relu(hn @ w1 + b1)
  dim3 gF(FFq / 128, Mrows / 128);
  gemm_wmma_kernel<0><<<gF, 128, 0, stream>>>(hn, w1t, b1, fh, nullptr, Mrows, FFq, Dq);

  // final tiny projection to [M,2]
  ffn2_kernel<<<Mrows / 8, 256, 0, stream>>>(fh, w2, b2, out);
}
